// BitNetMLP_87471303951008
// MI455X (gfx1250) — 8-bit matrix operand order repaired and checked by reading, compiled, not run on hardware
//
#include <hip/hip_runtime.h>

typedef int v8i __attribute__((ext_vector_type(8)));

#define H_DIM 512
#define I_DIM 2048
#define TOK_PER_BLK 16
#define XQ_STRIDE 528
#define HQ_STRIDE 2056

union APack { unsigned long long u[4]; v8i v; };
union BPack { int4 q[2]; v8i v; };

static __device__ __forceinline__ int quant8(float v, float s) {
  float q = rintf(v * s);
  q = fminf(fmaxf(q, -128.f), 127.f);
  return (int)q;
}


__global__ void wabs_partial(const float* __restrict__ wu,
                             const float* __restrict__ wd,
                             float* __restrict__ partials) {
  const int b = blockIdx.x;
  const float* w = (b < 128) ? wu : wd;
  const int bb = (b < 128) ? b : b - 128;
  const size_t base = (size_t)bb * 8192;
  float s = 0.f;
  for (int i = threadIdx.x; i < 8192; i += 256) s += fabsf(w[base + i]);
  __shared__ float red[256];
  red[threadIdx.x] = s;
  __syncthreads();
  for (int st = 128; st > 0; st >>= 1) {
    if (threadIdx.x < st) red[threadIdx.x] += red[threadIdx.x + st];
    __syncthreads();
  }
  if (threadIdx.x == 0) partials[b] = red[0];
}

__global__ void wscale_finalize(const float* __restrict__ partials,
                                float* __restrict__ scales) {
  if (threadIdx.x < 2) {
    const float* p = partials + threadIdx.x * 128;
    float s = 0.f;
    for (int i = 0; i < 128; i++) s += p[i];
    float mean = s * (1.f / 1048576.f);
    float clipped = fmaxf(mean, 1e-5f);
    scales[threadIdx.x * 2 + 0] = clipped;
    scales[threadIdx.x * 2 + 1] = 1.f / clipped;
  }
}

__global__ void wquant(const float* __restrict__ wu,
                       const float* __restrict__ wd,
                       const float* __restrict__ scales,
                       signed char* __restrict__ qu,
                       signed char* __restrict__ qd) {
  const size_t i = (size_t)blockIdx.x * 256 + threadIdx.x;
  const float su = scales[1], sd = scales[3];
  float a = rintf(wu[i] * su); a = fminf(fmaxf(a, -1.f), 1.f);
  float b = rintf(wd[i] * sd); b = fminf(fmaxf(b, -1.f), 1.f);
  qu[i] = (signed char)(int)a;
  qd[i] = (signed char)(int)b;
}


__global__ __launch_bounds__(256, 1)
void bitnet_mlp_fused(const float* __restrict__ x,
                      const float* __restrict__ g,
                      const signed char* __restrict__ wq_up,
                      const signed char* __restrict__ wq_dn,
                      const float* __restrict__ wsc,
                      float* __restrict__ out) {
  __shared__ __attribute__((aligned(16))) signed char xq[TOK_PER_BLK * XQ_STRIDE];
  __shared__ __attribute__((aligned(16))) signed char hq[TOK_PER_BLK * HQ_STRIDE];
  __shared__ float s_d1[16], s_rinv[16], s_asc2[16], s_d2[16];
  __shared__ float s_psum[8 * 16], s_pmax[8 * 16];

  const int tid = threadIdx.x;
  const int wv = tid >> 5;
  const int ln = tid & 31;
  const int lh = ln >> 4;
  const int l16 = ln & 15;
  const long tok_base = (long)blockIdx.x * TOK_PER_BLK;

  const float wmean_up = wsc[0];
  const float wmean_dn = wsc[2];

  const int tloc = 2 * wv + lh;
  const float4* xrow =
      reinterpret_cast<const float4*>(x + (size_t)(tok_base + tloc) * H_DIM) + l16 * 8;
  float4 xv[8];
  float amax = 0.f;
  #pragma unroll
  for (int c = 0; c < 8; c++) {
    xv[c] = xrow[c];
    amax = fmaxf(amax, fmaxf(fmaxf(fabsf(xv[c].x), fabsf(xv[c].y)),
                             fmaxf(fabsf(xv[c].z), fabsf(xv[c].w))));
  }
  #pragma unroll
  for (int m = 1; m < 16; m <<= 1) amax = fmaxf(amax, __shfl_xor(amax, m, 32));
  amax = fmaxf(amax, 1e-5f);
  const float asc1 = 127.f / amax;
  if (l16 == 0) s_d1[tloc] = (amax * (1.f / 127.f)) * wmean_up;
  int pk[8];
  #pragma unroll
  for (int c = 0; c < 8; c++) {
    unsigned q0 = (unsigned)quant8(xv[c].x, asc1) & 255u;
    unsigned q1 = (unsigned)quant8(xv[c].y, asc1) & 255u;
    unsigned q2 = (unsigned)quant8(xv[c].z, asc1) & 255u;
    unsigned q3 = (unsigned)quant8(xv[c].w, asc1) & 255u;
    pk[c] = (int)(q0 | (q1 << 8) | (q2 << 16) | (q3 << 24));
  }
  int4* xqd = reinterpret_cast<int4*>(&xq[tloc * XQ_STRIDE + l16 * 32]);
  xqd[0] = make_int4(pk[0], pk[1], pk[2], pk[3]);
  xqd[1] = make_int4(pk[4], pk[5], pk[6], pk[7]);
  __syncthreads();

  v8i A1[8];
  #pragma unroll
  for (int s = 0; s < 8; s++) {
    APack a;
    const unsigned long long* p =
        reinterpret_cast<const unsigned long long*>(&xq[l16 * XQ_STRIDE + s * 64 + lh * 16]);
    a.u[0] = p[0]; a.u[1] = p[1]; a.u[2] = p[4]; a.u[3] = p[5];
    A1[s] = a.v;
  }
  v8i acc1[16] = {};
  #pragma unroll
  for (int t = 0; t < 16; t++) {
    const signed char* brow = wq_up + (size_t)(256 * wv + 16 * t + l16) * H_DIM;
    #pragma unroll
    for (int s = 0; s < 8; s++) {
      BPack b;
      const int4* bp = reinterpret_cast<const int4*>(brow + s * 64 + lh * 16);
      b.q[0] = bp[0]; b.q[1] = bp[2];
      acc1[t] = __builtin_amdgcn_wmma_i32_16x16x64_iu8(true, A1[s], true, b.v,
                                                       acc1[t], false, false);
    }
  }

  float d1v[8];
  #pragma unroll
  for (int r = 0; r < 8; r++) d1v[r] = s_d1[r + 8 * lh];
  float gv[16], psum[8], pmax[8];
  #pragma unroll
  for (int r = 0; r < 8; r++) { psum[r] = 0.f; pmax[r] = 0.f; }
  #pragma unroll
  for (int t = 0; t < 16; t++) {
    gv[t] = g[256 * wv + 16 * t + l16];
    #pragma unroll
    for (int r = 0; r < 8; r++) {
      float hf = fmaxf((float)acc1[t][r] * d1v[r], 0.f);
      float h = hf * hf;
      psum[r] += h * h;
      pmax[r] = fmaxf(pmax[r], fabsf(h * gv[t]));
    }
  }
  #pragma unroll
  for (int m = 1; m < 16; m <<= 1) {
    #pragma unroll
    for (int r = 0; r < 8; r++) {
      psum[r] += __shfl_xor(psum[r], m, 32);
      pmax[r] = fmaxf(pmax[r], __shfl_xor(pmax[r], m, 32));
    }
  }
  if (l16 == 0) {
    #pragma unroll
    for (int r = 0; r < 8; r++) {
      s_psum[wv * 16 + 8 * lh + r] = psum[r];
      s_pmax[wv * 16 + 8 * lh + r] = pmax[r];
    }
  }
  __syncthreads();
  if (tid < 16) {
    float sum = 0.f, mx = 0.f;
    for (int w = 0; w < 8; w++) {
      sum += s_psum[w * 16 + tid];
      mx = fmaxf(mx, s_pmax[w * 16 + tid]);
    }
    float rinv = rsqrtf(sum * (1.f / 2048.f) + 1e-6f);
    float amax2 = fmaxf(rinv * mx, 1e-5f);
    s_rinv[tid] = rinv;
    s_asc2[tid] = 127.f / amax2;
    s_d2[tid] = (amax2 * (1.f / 127.f)) * wmean_dn;
  }
  __syncthreads();

  float coef[8];
  #pragma unroll
  for (int r = 0; r < 8; r++) {
    int m = r + 8 * lh;
    coef[r] = s_rinv[m] * s_asc2[m];
  }
  #pragma unroll
  for (int t = 0; t < 16; t++) {
    const int n = 256 * wv + 16 * t + l16;
    #pragma unroll
    for (int r = 0; r < 8; r++) {
      float hf = fmaxf((float)acc1[t][r] * d1v[r], 0.f);
      float h = hf * hf;
      hq[(r + 8 * lh) * HQ_STRIDE + n] = (signed char)quant8(h * gv[t], coef[r]);
    }
  }
  __syncthreads();

  v8i acc2[4] = {};
  #pragma unroll 8
  for (int s = 0; s < 32; s++) {
    APack a;
    const unsigned long long* p =
        reinterpret_cast<const unsigned long long*>(&hq[l16 * HQ_STRIDE + s * 64 + lh * 16]);
    a.u[0] = p[0]; a.u[1] = p[1]; a.u[2] = p[4]; a.u[3] = p[5];
    #pragma unroll
    for (int t = 0; t < 4; t++) {
      const signed char* brow = wq_dn + (size_t)(64 * wv + 16 * t + l16) * I_DIM;
      const int4* bp = reinterpret_cast<const int4*>(brow + s * 64 + lh * 16);
      BPack b;
      b.q[0] = bp[0]; b.q[1] = bp[2];
      acc2[t] = __builtin_amdgcn_wmma_i32_16x16x64_iu8(true, a.v, true, b.v,
                                                       acc2[t], false, false);
    }
  }

  float d2v[8];
  #pragma unroll
  for (int r = 0; r < 8; r++) d2v[r] = s_d2[r + 8 * lh];
  #pragma unroll
  for (int t = 0; t < 4; t++) {
    const int n = 64 * wv + 16 * t + l16;
    #pragma unroll
    for (int r = 0; r < 8; r++) {
      out[(size_t)(tok_base + r + 8 * lh) * H_DIM + n] = (float)acc2[t][r] * d2v[r];
    }
  }
}


extern "C" void kernel_launch(void* const* d_in, const int* in_sizes, int n_in,
                              void* d_out, int out_size, void* d_ws, size_t ws_size,
                              hipStream_t stream) {
  const float* x      = (const float*)d_in[0];
  const float* w_up   = (const float*)d_in[1];
  const float* w_down = (const float*)d_in[2];
  const float* g      = (const float*)d_in[3];
  float* out = (float*)d_out;

  char* ws = (char*)d_ws;
  signed char* qu = (signed char*)ws;
  signed char* qd = qu + (1u << 20);
  float* partials = (float*)(ws + (2u << 20));
  float* scales = partials + 256;

  const int ntok = in_sizes[0] / H_DIM;
  const int nblk = ntok / TOK_PER_BLK;

  wabs_partial<<<256, 256, 0, stream>>>(w_up, w_down, partials);
  wscale_finalize<<<1, 64, 0, stream>>>(partials, scales);
  wquant<<<4096, 256, 0, stream>>>(w_up, w_down, scales, qu, qd);
  bitnet_mlp_fused<<<nblk, 256, 0, stream>>>(x, g, qu, qd, scales, out);
  (void)n_in; (void)out_size; (void)ws_size;
}
